// ConditioningGNN_85727547228372
// MI455X (gfx1250) — matrix operand order repaired and checked by reading, not compiled, not run on hardware
//
#include <hip/hip_runtime.h>
#include <hip/hip_bf16.h>

// ---------------------------------------------------------------------------
// ConditioningGNN for MI455X (gfx1250, wave32, WMMA)
//
// Pipeline (all on `stream`, ping-pong buffers A/B in d_ws):
//   1. deg  = 1 (self loop) + atomic count of dst          -> dinv = rsqrt
//   2. hw1  = concat(x, y[batch]) @ W1        (K=6, scalar) -> bufA
//   3. agg1 = hw1*dinv^2 (self loop preset)                 -> bufB
//      agg1 += scatter_e( hw1[src]*dinv[src]*dinv[dst] )    (atomics)
//      h1   = relu(agg1 + b1)                                (in place, bufB)
//   4. hw2  = h1 @ W2    (WMMA f16->f32, 16x16x32 tiles)    -> bufA
//   5. agg2/h2 same as step 3 with b2                       -> bufB
//   6. g    = relu(h2 @ W3 + b3)   (WMMA, fused bias+relu)  -> bufA
//   7. st   = g @ W4 + b4 (N=8, scalar), split-store s|t    -> d_out
// ---------------------------------------------------------------------------

typedef __attribute__((ext_vector_type(16))) _Float16 v16h;
typedef __attribute__((ext_vector_type(8)))  _Float16 v8h;
typedef __attribute__((ext_vector_type(8)))  float    v8f;

#define HID 128
#define LDS_STRIDE 136   // halves; 272B rows -> every K-chunk offset is 16B aligned
#define TILE_ROWS 128    // node rows per block in the WMMA GEMM

// ---------------- small elementwise / scalar kernels -----------------------

__global__ void init_deg_kernel(float* deg, int n) {
    int i = blockIdx.x * blockDim.x + threadIdx.x;
    if (i < n) deg[i] = 1.0f;   // self-loop
}

__global__ void count_deg_kernel(const long long* __restrict__ dst,
                                 float* __restrict__ deg, int n_edges) {
    int e = blockIdx.x * blockDim.x + threadIdx.x;
    if (e < n_edges) atomicAdd(&deg[dst[e]], 1.0f);
}

__global__ void rsqrt_kernel(float* deg, int n) {
    int i = blockIdx.x * blockDim.x + threadIdx.x;
    if (i < n) deg[i] = rsqrtf(deg[i]);
}

// hw1 = concat(x, y[batch]) @ W1   (K = 6: 4 from x, 2 from y)
__global__ void gemm1_kernel(const float* __restrict__ x,
                             const float* __restrict__ y,
                             const long long* __restrict__ batch,
                             const float* __restrict__ W1,
                             float* __restrict__ out, int n_nodes) {
    int gid = blockIdx.x * blockDim.x + threadIdx.x;
    if (gid >= n_nodes * HID) return;
    int n = gid >> 7, c = gid & 127;
    long long b = batch[n];
    float acc = 0.0f;
    acc += x[(size_t)n * 4 + 0] * W1[0 * HID + c];
    acc += x[(size_t)n * 4 + 1] * W1[1 * HID + c];
    acc += x[(size_t)n * 4 + 2] * W1[2 * HID + c];
    acc += x[(size_t)n * 4 + 3] * W1[3 * HID + c];
    acc += y[(size_t)b * 2 + 0] * W1[4 * HID + c];
    acc += y[(size_t)b * 2 + 1] * W1[5 * HID + c];
    out[gid] = acc;
}

// agg = hw * dinv[n]^2   (self-loop contribution, also initializes agg)
__global__ void preset_kernel(const float* __restrict__ hw,
                              const float* __restrict__ dinv,
                              float* __restrict__ agg, int n_nodes) {
    int gid = blockIdx.x * blockDim.x + threadIdx.x;
    if (gid >= n_nodes * HID) return;
    float di = dinv[gid >> 7];
    agg[gid] = hw[gid] * di * di;
}

// edge scatter: 16 threads per edge x 8 channels each (512B coalesced)
__global__ void scatter_kernel(const long long* __restrict__ src,
                               const long long* __restrict__ dst,
                               const float* __restrict__ dinv,
                               const float* __restrict__ hw,
                               float* __restrict__ agg, int n_edges) {
    long long gid = (long long)blockIdx.x * blockDim.x + threadIdx.x;
    if (gid >= (long long)n_edges * 16) return;
    long long e = gid >> 4;
    int c0 = (int)(gid & 15) * 8;
    long long s = src[e], d = dst[e];
    float norm = dinv[s] * dinv[d];
    const float* hs = hw + (size_t)s * HID + c0;
    float* ad = agg + (size_t)d * HID + c0;
#pragma unroll
    for (int j = 0; j < 8; ++j)
        atomicAdd(ad + j, hs[j] * norm);
}

__global__ void bias_relu_kernel(float* __restrict__ h,
                                 const float* __restrict__ bias, int n_nodes) {
    int gid = blockIdx.x * blockDim.x + threadIdx.x;
    if (gid >= n_nodes * HID) return;
    h[gid] = fmaxf(h[gid] + bias[gid & 127], 0.0f);
}

// st = g @ W4 + b4, split-store: s -> out[0:4N), t -> out[4N:8N)
__global__ void gemm4_kernel(const float* __restrict__ g,
                             const float* __restrict__ W4,
                             const float* __restrict__ b4,
                             float* __restrict__ out, int n_nodes) {
    int gid = blockIdx.x * blockDim.x + threadIdx.x;
    if (gid >= n_nodes * 8) return;
    int n = gid >> 3, c = gid & 7;
    const float* gr = g + (size_t)n * HID;
    float acc = b4[c];
#pragma unroll 16
    for (int k = 0; k < HID; ++k)
        acc += gr[k] * W4[k * 8 + c];
    if (c < 4) out[(size_t)n * 4 + c] = acc;
    else       out[(size_t)n_nodes * 4 + (size_t)n * 4 + (c - 4)] = acc;
}

// ---------------- WMMA GEMM: Out[N,128] = H[N,128] @ W[128,128] -------------
// 256 threads = 8 waves; each wave owns a 16-row m-tile of the 128-row block,
// computes eight 16x16 n-tiles via 4 k-steps of v_wmma_f32_16x16x32_f16.
// W is staged TRANSPOSED in LDS so every A and B fragment is built from
// contiguous, 16B-aligned 8-half chunks (wide ds_load_b128, no packing).
// Bias/ReLU are compile-time template params; full tiles take a uniform
// branch-free store path (only the final partial block runs row guards).
template <bool HAS_BIAS, bool RELU>
__global__ __launch_bounds__(256) void gemm128_wmma_kernel(
    const float* __restrict__ H, const float* __restrict__ W,
    const float* __restrict__ bias, float* __restrict__ Out,
    int n_rows) {
    __shared__ __align__(16) _Float16 Wt[HID * LDS_STRIDE];       // Wt[col][k]
    __shared__ __align__(16) _Float16 Hl[TILE_ROWS * LDS_STRIDE]; // Hl[row][k]

    const int tid = threadIdx.x;

    // Stage W transposed (W[k][c] -> Wt[c][k]): 64 elements per thread
    for (int i = tid; i < HID * HID; i += 256) {
        int k = i >> 7, c = i & 127;
        Wt[c * LDS_STRIDE + k] = (_Float16)W[i];
    }
    // Stage this block's 128 H rows (zero-padded past n_rows)
    const int row0 = blockIdx.x * TILE_ROWS;
    for (int i = tid; i < TILE_ROWS * HID; i += 256) {
        int r = i >> 7, c = i & 127;
        int gr = row0 + r;
        float v = (gr < n_rows) ? H[(size_t)gr * HID + c] : 0.0f;
        Hl[r * LDS_STRIDE + c] = (_Float16)v;
    }
    __syncthreads();

    const int wave = tid >> 5;
    const int lane = tid & 31;
    const int half = lane >> 4;   // selects K sub-range in A/B frags, M group in D
    const int l16  = lane & 15;
    const int mrow = wave * 16 + l16;   // A-matrix row (lane layout: M = lane%16)

    union Frag { v16h v; v8h h[2]; };

    // A fragments for the 4 k-steps (CDNA5 16-bit A layout):
    //   elements 0..7  = K = 32*kk + half*8 + (0..7)       (contiguous chunk)
    //   elements 8..15 = K = 32*kk + 16 + half*8 + (0..7)  (contiguous chunk)
    v16h afrag[4];
#pragma unroll
    for (int kk = 0; kk < 4; ++kk) {
        const _Float16* arow = &Hl[mrow * LDS_STRIDE + 32 * kk + half * 8];
        Frag a;
        a.h[0] = *(const v8h*)(arow);        // K offsets 0..7
        a.h[1] = *(const v8h*)(arow + 16);   // K offsets 16..23
        afrag[kk] = a.v;
    }

    // D-matrix row base for this lane: lanes 0-15 -> M=0..7, 16-31 -> M=8..15
    const int drow0 = row0 + wave * 16 + half * 8;
    const bool full = (row0 + TILE_ROWS) <= n_rows;   // uniform across block

#pragma unroll
    for (int nt = 0; nt < 8; ++nt) {
        v8f acc = {};
#pragma unroll
        for (int kk = 0; kk < 4; ++kk) {
            // B fragment (32x16, K x N): lane holds col = lane%16,
            // elements 0..7 = K = 32*kk + half*8 + (0..7), 8..15 = K = 32*kk + 16 + half*8 + (0..7)
            const _Float16* bcol =
                &Wt[(nt * 16 + l16) * LDS_STRIDE + 32 * kk + half * 8];
            Frag b;
            b.h[0] = *(const v8h*)(bcol);
            b.h[1] = *(const v8h*)(bcol + 16);
            acc = __builtin_amdgcn_wmma_f32_16x16x32_f16(
                false, afrag[kk], false, b.v, (short)0, acc, false, false);
        }

        const int col = nt * 16 + l16;
        float bval = 0.0f;
        if (HAS_BIAS) bval = bias[col];           // one coalesced load per n-tile
        float* op = Out + (size_t)drow0 * HID + col;

        if (full) {                               // uniform branch: 781/782 blocks
#pragma unroll
            for (int r = 0; r < 8; ++r) {
                float v = acc[r];
                if (HAS_BIAS) v += bval;
                if (RELU) v = fmaxf(v, 0.0f);
                op[(size_t)r * HID] = v;
            }
        } else {
#pragma unroll
            for (int r = 0; r < 8; ++r) {
                if (drow0 + r < n_rows) {
                    float v = acc[r];
                    if (HAS_BIAS) v += bval;
                    if (RELU) v = fmaxf(v, 0.0f);
                    op[(size_t)r * HID] = v;
                }
            }
        }
    }
}

// ---------------------------------------------------------------------------

extern "C" void kernel_launch(void* const* d_in, const int* in_sizes, int n_in,
                              void* d_out, int out_size, void* d_ws, size_t ws_size,
                              hipStream_t stream) {
    const float*     x     = (const float*)d_in[0];
    const long long* edge  = (const long long*)d_in[1];   // [2, E] int64
    const float*     y     = (const float*)d_in[2];
    const long long* batch = (const long long*)d_in[3];
    const float*     W1    = (const float*)d_in[4];
    const float*     b1    = (const float*)d_in[5];
    const float*     W2    = (const float*)d_in[6];
    const float*     b2    = (const float*)d_in[7];
    const float*     W3    = (const float*)d_in[8];
    const float*     b3    = (const float*)d_in[9];
    const float*     W4    = (const float*)d_in[10];
    const float*     b4    = (const float*)d_in[11];

    const int N = in_sizes[0] / 4;   // 100000
    const int E = in_sizes[1] / 2;   // 1600000
    const long long* src = edge;
    const long long* dst = edge + E;

    // Workspace layout: dinv | bufA | bufB
    char* ws = (char*)d_ws;
    float* dinv = (float*)ws;
    size_t off  = (((size_t)N * 4) + 255) & ~(size_t)255;
    float* bufA = (float*)(ws + off);
    float* bufB = bufA + (size_t)N * HID;
    (void)n_in; (void)out_size; (void)ws_size;

    const int T = 256;
    const int gN    = (N + T - 1) / T;
    const int gE    = (E + T - 1) / T;
    const int gNH   = (N * HID + T - 1) / T;
    const int gE16  = (int)(((long long)E * 16 + T - 1) / T);
    const int gTile = (N + TILE_ROWS - 1) / TILE_ROWS;
    const int gN8   = (N * 8 + T - 1) / T;

    float* out = (float*)d_out;

    // 1. degrees & norm
    init_deg_kernel<<<gN, T, 0, stream>>>(dinv, N);
    count_deg_kernel<<<gE, T, 0, stream>>>(dst, dinv, E);
    rsqrt_kernel<<<gN, T, 0, stream>>>(dinv, N);

    // 2. layer 1 dense transform (K=6, scalar)
    gemm1_kernel<<<gNH, T, 0, stream>>>(x, y, batch, W1, bufA, N);

    // 3. layer 1 aggregate: self-loop preset + edge scatter + bias/relu
    preset_kernel<<<gNH, T, 0, stream>>>(bufA, dinv, bufB, N);
    scatter_kernel<<<gE16, T, 0, stream>>>(src, dst, dinv, bufA, bufB, E);
    bias_relu_kernel<<<gNH, T, 0, stream>>>(bufB, b1, N);

    // 4. layer 2 dense transform (WMMA, plain)
    gemm128_wmma_kernel<false, false><<<gTile, T, 0, stream>>>(bufB, W2, nullptr, bufA, N);

    // 5. layer 2 aggregate
    preset_kernel<<<gNH, T, 0, stream>>>(bufA, dinv, bufB, N);
    scatter_kernel<<<gE16, T, 0, stream>>>(src, dst, dinv, bufA, bufB, E);
    bias_relu_kernel<<<gNH, T, 0, stream>>>(bufB, b2, N);

    // 6. MLP hidden layer (WMMA, fused bias+relu)
    gemm128_wmma_kernel<true, true><<<gTile, T, 0, stream>>>(bufB, W3, b3, bufA, N);

    // 7. output head + split store
    gemm4_kernel<<<gN8, T, 0, stream>>>(bufA, W4, b4, out, N);
}
